// SubstructureAwarePooling_26285199851683
// MI455X (gfx1250) — matrix operand order repaired and checked by reading, not compiled, not run on hardware
//
#include <hip/hip_runtime.h>
#include <hip/hip_bf16.h>
#include <math.h>
#include <stdint.h>

typedef __attribute__((ext_vector_type(8)))  float  v8f;
typedef __attribute__((ext_vector_type(16))) __bf16 v16bf;
typedef __attribute__((ext_vector_type(8)))  __bf16 v8bf;
typedef __attribute__((ext_vector_type(4)))  unsigned int u32x4;
typedef __attribute__((ext_vector_type(8)))  int i32x8;
typedef __attribute__((ext_vector_type(4)))  int i32x4;

// ---------------- WMMA helpers (CDNA5, wave32) ----------------

__device__ __forceinline__ v8f wmma_bf16(v16bf a, v16bf b, v8f c) {
  // v_wmma_f32_16x16x32_bf16, D = A*B + C, fp32 accumulate
  return __builtin_amdgcn_wmma_f32_16x16x32_bf16(false, a, false, b, (short)0, c,
                                                 false, false);
}

// A fragment: 16x32 bf16, row-major in LDS with stride lda.
// ISA layout: lanes 0-15 hold M=lane, K 0..7 & 16..23; lanes 16-31 hold K 8..15 & 24..31.
__device__ __forceinline__ v16bf load_a_frag(const __bf16* A, int lda, int k0, int lane) {
  int m  = lane & 15;
  int kb = (lane >> 4) << 3;          // 0 or 8
  const __bf16* p = A + m * lda + k0 + kb;
  v8bf lo = *(const v8bf*)(p);        // K = kb .. kb+7
  v8bf hi = *(const v8bf*)(p + 16);   // K = kb+16 .. kb+23
  return __builtin_shufflevector(lo, hi, 0,1,2,3,4,5,6,7,8,9,10,11,12,13,14,15);
}

// Same fragment but source is fp32 in LDS (convert in registers).
__device__ __forceinline__ v16bf load_a_frag_f32(const float* A, int lda, int k0,
                                                 int lane) {
  int m  = lane & 15;
  int kb = (lane >> 4) << 3;
  const float* p = A + m * lda + k0 + kb;
  v16bf r;
  #pragma unroll
  for (int j = 0; j < 8; ++j) r[j] = (__bf16)p[j];
  #pragma unroll
  for (int j = 0; j < 8; ++j) r[8 + j] = (__bf16)p[16 + j];
  return r;
}

// B fragments pre-swizzled in workspace: tile (nt,kt) stored as [lane][16 elems].
__device__ __forceinline__ v16bf load_b_frag(const __bf16* Wp, int KT, int kt, int nt,
                                             int lane) {
  return *(const v16bf*)(Wp + (((size_t)(nt * KT + kt) * 32 + lane) << 4));
}

// TDM: async DMA of `elems` fp32 from global `src` into LDS at byte offset `lds`.
// D# group0: count=1 | lds_addr | global_addr | type=2. group1: data_size=4B,
// tensor_dim0=tile_dim0=elems (1 row), tensor_dim1=1, tile_dim1=1.
__device__ __forceinline__ void tdm_load_f32(const float* src, unsigned lds,
                                             unsigned elems) {
  unsigned long long ga = (unsigned long long)(uintptr_t)src;
  u32x4 g0;
  g0.x = 1u;                                    // count = 1 (valid user D#)
  g0.y = lds;                                   // lds_addr (bytes)
  g0.z = (unsigned)ga;                          // global_addr[31:0]
  g0.w = (unsigned)(ga >> 32) | 0x80000000u;    // global_addr[56:32] | type=2
  i32x8 g1;
  g1[0] = 0x00020000;                           // data_size = 2 (4 bytes)
  g1[1] = (int)(elems << 16);                   // tensor_dim0[15:0]
  g1[2] = (int)((elems >> 16) | (1u << 16));    // tensor_dim0[31:16] | tensor_dim1=1
  g1[3] = (int)(elems << 16);                   // tile_dim0 = elems (<=65535)
  g1[4] = 1;                                    // tile_dim1 = 1, tile_dim2 = 0
  g1[5] = (int)elems;                           // tensor_dim0_stride
  g1[6] = 0;
  g1[7] = 0;
  i32x4 gz = {0, 0, 0, 0};
  asm volatile("tensor_load_to_lds %0, %1, %2, %3"
               :: "s"(g0), "s"(g1), "s"(gz), "s"(gz)
               : "memory");
}

// ---------------- Kernels ----------------

__global__ void k_zero(float* p, int n) {
  int i = blockIdx.x * blockDim.x + threadIdx.x;
  for (; i < n; i += gridDim.x * blockDim.x) p[i] = 0.f;
}

// Swizzle fp32 weight [K,128] row-major -> bf16 WMMA B-fragment layout.
__global__ void k_swz(const float* __restrict__ W, __bf16* __restrict__ dst, int KT) {
  int t = blockIdx.x * blockDim.x + threadIdx.x;
  int total = KT * 8 * 512;
  if (t >= total) return;
  int e = t & 15;
  int l = (t >> 4) & 31;
  int tile = t >> 9;
  int kt = tile % KT, nt = tile / KT;
  int n = nt * 16 + (l & 15);
  int k = kt * 32 + ((l >> 4) << 3) + (e & 7) + ((e >> 3) << 4);
  dst[t] = (__bf16)W[k * 128 + n];
}

// Per-graph sum(h), sum(h^2), counts. batch is sorted -> run-length accumulation.
__global__ void k_gstats(const float* __restrict__ h, const int* __restrict__ batch,
                         float* gsum, float* gsq, float* counts, int N) {
  __shared__ int sb[64];
  int base = blockIdx.x * 64;
  int d = threadIdx.x;  // 128 threads = columns
  if (d < 64) { int r = base + d; sb[d] = (r < N) ? batch[r] : -1; }
  __syncthreads();
  int lim = N - base; if (lim > 64) lim = 64;
  if (lim <= 0) return;
  float s = 0.f, q = 0.f;
  int gc = sb[0];
  for (int r = 0; r < lim; ++r) {
    int g = sb[r];
    float x = h[(size_t)(base + r) * 128 + d];
    if (g != gc) {
      atomicAdd(&gsum[gc * 128 + d], s);
      atomicAdd(&gsq [gc * 128 + d], q);
      s = 0.f; q = 0.f; gc = g;
    }
    s += x; q += x * x;
  }
  atomicAdd(&gsum[gc * 128 + d], s);
  atomicAdd(&gsq [gc * 128 + d], q);
  if (d == 0) {
    int c = 0, g2 = sb[0];
    for (int r = 0; r < lim; ++r) {
      int g = sb[r];
      if (g != g2) { atomicAdd(&counts[g2], (float)c); c = 0; g2 = g; }
      ++c;
    }
    if (c > 0) atomicAdd(&counts[g2], (float)c);
  }
}

// GEMM1: Y1 = sub[N,64] @ W1 + b1, bf16 out, accumulate BN1 column sums.
__global__ void __launch_bounds__(256) k_gemm1(
    const float* __restrict__ sub, const __bf16* __restrict__ Wp1,
    const float* __restrict__ b1, __bf16* __restrict__ y1,
    float* bn1_sum, float* bn1_sq, int NT) {
  __shared__ float  sh_a[16 * 64];
  __shared__ __bf16 sh_y[16 * 128];
  int lane = threadIdx.x & 31, nt = threadIdx.x >> 5;
  v16bf B0 = load_b_frag(Wp1, 2, 0, nt, lane);
  v16bf B1 = load_b_frag(Wp1, 2, 1, nt, lane);
  int ncol = nt * 16 + (lane & 15);
  int mbase = (lane >> 4) << 3;
  float bias = b1[ncol];
  float cs = 0.f, cq = 0.f;
  for (int mt = blockIdx.x; mt < NT; mt += gridDim.x) {
    size_t rbase = (size_t)mt * 16;
    // TDM: wave 0 DMAs the 16x64 fp32 sub tile into LDS.
    if (nt == 0)
      tdm_load_f32(sub + rbase * 64, (unsigned)(uintptr_t)(void*)sh_a, 1024u);
    if (nt == 0) __builtin_amdgcn_s_wait_tensorcnt(0);
    __syncthreads();
    v8f c = {0.f,0.f,0.f,0.f,0.f,0.f,0.f,0.f};
    c = wmma_bf16(load_a_frag_f32(sh_a, 64,  0, lane), B0, c);
    c = wmma_bf16(load_a_frag_f32(sh_a, 64, 32, lane), B1, c);
    #pragma unroll
    for (int r = 0; r < 8; ++r) {
      float y = c[r] + bias;
      cs += y; cq += y * y;
      sh_y[(mbase + r) * 128 + ncol] = (__bf16)y;
    }
    __syncthreads();
    {  // coalesced bf16 store via LDS staging
      int t = threadIdx.x;
      v8bf v = ((const v8bf*)sh_y)[t];
      ((v8bf*)(y1 + rbase * 128))[t] = v;
    }
    __syncthreads();
  }
  atomicAdd(&bn1_sum[ncol], cs);
  atomicAdd(&bn1_sq [ncol], cq);
}

// Finalize BN1 scale/shift, graph mean / inv-(std+eps) in place, segment offsets.
__global__ void k_fin1(const float* g1, const float* be1,
                       const float* b1s, const float* b1q,
                       float* sc1, float* sh1,
                       float* gsum, float* gsq, const float* counts,
                       int* segstart, float fN) {
  int t = threadIdx.x;
  if (t < 128) {
    float mean = b1s[t] / fN;
    float var  = b1q[t] / fN - mean * mean;
    float rstd = rsqrtf(var + 1e-5f);
    float s = g1[t] * rstd;
    sc1[t] = s;
    sh1[t] = be1[t] - mean * s;
  }
  for (int i = t; i < 256 * 128; i += blockDim.x) {
    int g = i >> 7;
    float cnt = fmaxf(counts[g], 1.0f);
    float m = gsum[i] / cnt;
    float var = gsq[i] / cnt - m * m;
    float sd = sqrtf(fmaxf(var, 1e-8f));
    gsum[i] = m;
    gsq[i]  = 1.0f / (sd + 1e-8f);
  }
  if (t == 0) {
    int acc = 0;
    for (int g = 0; g < 256; ++g) { segstart[g] = acc; acc += (int)(counts[g] + 0.5f); }
    segstart[256] = acc;
  }
}

// Mega-fused pass: BN1+ReLU -> gate GEMM -> sigmoid -> h_fused -> h_dev -> score GEMM.
__global__ void __launch_bounds__(256) k_fused(
    const float* __restrict__ h, const int* __restrict__ batch,
    __bf16* __restrict__ y1z,       // in: Y1 pre-BN; out (overwritten): Z pre-BN
    __bf16* __restrict__ hfused,
    const __bf16* __restrict__ Wpg, const __bf16* __restrict__ Wps1,
    const float* __restrict__ bg, const float* __restrict__ bs1,
    const float* __restrict__ sc1, const float* __restrict__ sh1,
    const float* __restrict__ gmean, const float* __restrict__ ginv,
    float* bn2_sum, float* bn2_sq, int NT) {
  __shared__ float  sh_h[16 * 128];   // h tile fp32 (filled by TDM)
  __shared__ __bf16 sh_s[16 * 128];   // sub_enh, later h_dev
  __shared__ __bf16 sh_f[16 * 128];   // h_fused bf16
  __shared__ __bf16 sh_z[16 * 128];   // Z pre-BN staging
  __shared__ int    sh_b[16];
  int lane = threadIdx.x & 31, nt = threadIdx.x >> 5;
  v16bf Bg[8], Bs[8];
  #pragma unroll
  for (int kt = 0; kt < 8; ++kt) {
    Bg[kt] = load_b_frag(Wpg,  8, kt, nt, lane);
    Bs[kt] = load_b_frag(Wps1, 8, kt, nt, lane);
  }
  int ncol = nt * 16 + (lane & 15);
  int mbase = (lane >> 4) << 3;
  float bgv = bg[ncol], bsv = bs1[ncol];
  float cs = 0.f, cq = 0.f;
  for (int mt = blockIdx.x; mt < NT; mt += gridDim.x) {
    size_t rbase = (size_t)mt * 16;
    // Wave 0: async TDM copy of the 16x128 fp32 h tile -> sh_h (no VGPR round-trip).
    if (nt == 0)
      tdm_load_f32(h + rbase * 128, (unsigned)(uintptr_t)(void*)sh_h, 2048u);
    {  // Meanwhile: Y1 tile -> BN1 + ReLU -> sub_enh bf16 (overlaps with TDM)
      int t = threadIdx.x;
      v8bf yv = ((const v8bf*)(y1z + rbase * 128))[t];
      int li = t * 8, col = li & 127;
      __bf16* pd = sh_s + li;
      #pragma unroll
      for (int e = 0; e < 8; ++e) {
        float x = (float)yv[e] * sc1[col + e] + sh1[col + e];
        pd[e] = (__bf16)fmaxf(x, 0.f);
      }
    }
    if (threadIdx.x < 16) sh_b[threadIdx.x] = batch[rbase + threadIdx.x];
    if (mt + gridDim.x < NT)
      __builtin_prefetch(h + (rbase + (size_t)gridDim.x * 16) * 128 + threadIdx.x * 8, 0, 0);
    if (nt == 0) __builtin_amdgcn_s_wait_tensorcnt(0);
    __syncthreads();  // B1: h tile (TDM) + sub_enh ready
    // Gate GEMM: A = [h (fp32, cvt in regs) | sub_enh (bf16)]
    v8f acc = {0.f,0.f,0.f,0.f,0.f,0.f,0.f,0.f};
    #pragma unroll
    for (int kt = 0; kt < 4; ++kt)
      acc = wmma_bf16(load_a_frag_f32(sh_h, 128, kt * 32, lane), Bg[kt], acc);
    #pragma unroll
    for (int kt = 4; kt < 8; ++kt)
      acc = wmma_bf16(load_a_frag(sh_s, 128, (kt - 4) * 32, lane), Bg[kt], acc);
    // sigmoid + h_fused (writes only sh_f -> no barrier needed before this)
    #pragma unroll
    for (int r = 0; r < 8; ++r) {
      int m = mbase + r;
      float gate = 1.f / (1.f + __expf(-(acc[r] + bgv)));
      float hf = sh_h[m * 128 + ncol] + gate * (float)sh_s[m * 128 + ncol];
      sh_f[m * 128 + ncol] = (__bf16)hf;
    }
    __syncthreads();  // B2: all reads of sh_s done; sh_f complete
    #pragma unroll
    for (int it = 0; it < 8; ++it) {  // h_dev overwrites sh_s
      int t = threadIdx.x + it * 256;
      int row = t >> 7, col = t & 127, g = sh_b[row];
      float dv = (sh_h[t] - gmean[g * 128 + col]) * ginv[g * 128 + col];
      sh_s[t] = (__bf16)dv;
    }
    __syncthreads();  // B3
    // Score GEMM: A = [h_fused | h_dev]
    v8f acc2 = {0.f,0.f,0.f,0.f,0.f,0.f,0.f,0.f};
    #pragma unroll
    for (int kt = 0; kt < 4; ++kt)
      acc2 = wmma_bf16(load_a_frag(sh_f, 128, kt * 32, lane), Bs[kt], acc2);
    #pragma unroll
    for (int kt = 4; kt < 8; ++kt)
      acc2 = wmma_bf16(load_a_frag(sh_s, 128, (kt - 4) * 32, lane), Bs[kt], acc2);
    #pragma unroll
    for (int r = 0; r < 8; ++r) {
      float zp = acc2[r] + bsv;
      cs += zp; cq += zp * zp;
      sh_z[(mbase + r) * 128 + ncol] = (__bf16)zp;
    }
    __syncthreads();  // B4
    {  // coalesced stores: h_fused and Z pre-BN (overwrites Y1 region)
      int t = threadIdx.x;
      v8bf hv = ((const v8bf*)sh_f)[t];
      ((v8bf*)(hfused + rbase * 128))[t] = hv;
      v8bf zv = ((const v8bf*)sh_z)[t];
      ((v8bf*)(y1z + rbase * 128))[t] = zv;
    }
    __syncthreads();  // B5: before next iteration's TDM/staging overwrites
  }
  atomicAdd(&bn2_sum[ncol], cs);
  atomicAdd(&bn2_sq [ncol], cq);
}

__global__ void k_fin2(const float* g2, const float* be2, const float* s, const float* q,
                       float* sc2, float* sh2, float fN) {
  int t = threadIdx.x;
  if (t < 128) {
    float mean = s[t] / fN;
    float var  = q[t] / fN - mean * mean;
    float rstd = rsqrtf(var + 1e-5f);
    float sc = g2[t] * rstd;
    sc2[t] = sc;
    sh2[t] = be2[t] - mean * sc;
  }
}

// logits = relu(BN2(Z)) @ Ws2 + bs2  (wave-per-row GEMV, shfl_xor reduce)
__global__ void __launch_bounds__(256) k_logits(
    const __bf16* __restrict__ zbuf, const float* __restrict__ sc2,
    const float* __restrict__ sh2, const float* __restrict__ Ws2,
    const float* __restrict__ bs2, float* __restrict__ logits, int N) {
  int lane = threadIdx.x & 31;
  int wid = (blockIdx.x * blockDim.x + threadIdx.x) >> 5;
  int tw  = (gridDim.x * blockDim.x) >> 5;
  for (int row = wid; row < N; row += tw) {
    float p = 0.f;
    #pragma unroll
    for (int j = 0; j < 4; ++j) {
      int c = lane + j * 32;
      float z = fmaxf((float)zbuf[(size_t)row * 128 + c] * sc2[c] + sh2[c], 0.f);
      p += z * Ws2[c];
    }
    p += __shfl_xor(p, 16); p += __shfl_xor(p, 8); p += __shfl_xor(p, 4);
    p += __shfl_xor(p, 2);  p += __shfl_xor(p, 1);
    if (lane == 0) logits[row] = p + bs2[0];
  }
}

// Per-segment max and exp-sum (segments are contiguous since batch is sorted).
__global__ void k_seg(const float* __restrict__ logits, const int* __restrict__ segstart,
                      float* segmax, float* segden) {
  __shared__ float red[256];
  int g = blockIdx.x, t = threadIdx.x;
  int s = segstart[g], e = segstart[g + 1];
  float m = -3.0e38f;
  for (int i = s + t; i < e; i += 256) m = fmaxf(m, logits[i]);
  red[t] = m; __syncthreads();
  for (int o = 128; o > 0; o >>= 1) { if (t < o) red[t] = fmaxf(red[t], red[t + o]); __syncthreads(); }
  float mx = red[0]; __syncthreads();
  float sum = 0.f;
  for (int i = s + t; i < e; i += 256) sum += __expf(logits[i] - mx);
  red[t] = sum; __syncthreads();
  for (int o = 128; o > 0; o >>= 1) { if (t < o) red[t] += red[t + o]; __syncthreads(); }
  if (t == 0) { segmax[g] = mx; segden[g] = (red[0] > 0.f) ? red[0] : 1.0f; }
}

// out[g] = alpha * sum(h_fused * score) + (1-alpha) * graph_mean
__global__ void k_out(const __bf16* __restrict__ hfused, const float* __restrict__ logits,
                      const int* __restrict__ segstart, const float* __restrict__ segmax,
                      const float* __restrict__ segden, const float* __restrict__ gmean,
                      const float* __restrict__ mix, float* __restrict__ out) {
  __shared__ float ssc[128];
  int g = blockIdx.x, d = threadIdx.x;
  int s = segstart[g], e = segstart[g + 1];
  float mx = segmax[g], dn = segden[g];
  float acc = 0.f;
  for (int base = s; base < e; base += 128) {
    int i = base + d;
    ssc[d] = (i < e) ? __expf(logits[i] - mx) / dn : 0.f;
    __syncthreads();
    int cnt = e - base; if (cnt > 128) cnt = 128;
    for (int r = 0; r < cnt; ++r)
      acc += (float)hfused[(size_t)(base + r) * 128 + d] * ssc[r];
    __syncthreads();
  }
  float alpha = 1.f / (1.f + __expf(-mix[0]));
  out[g * 128 + d] = alpha * acc + (1.f - alpha) * gmean[g * 128 + d];
}

// ---------------- Host launcher ----------------

extern "C" void kernel_launch(void* const* d_in, const int* in_sizes, int n_in,
                              void* d_out, int out_size, void* d_ws, size_t ws_size,
                              hipStream_t stream) {
  (void)n_in; (void)out_size; (void)ws_size;
  const float* h    = (const float*)d_in[0];
  const float* sub  = (const float*)d_in[1];
  const int*   batch= (const int*)  d_in[2];
  const float* W1   = (const float*)d_in[3];
  const float* b1   = (const float*)d_in[4];
  const float* g1   = (const float*)d_in[5];
  const float* be1  = (const float*)d_in[6];
  const float* Wg   = (const float*)d_in[7];
  const float* bg   = (const float*)d_in[8];
  const float* Ws1  = (const float*)d_in[9];
  const float* bs1  = (const float*)d_in[10];
  const float* g2   = (const float*)d_in[11];
  const float* be2  = (const float*)d_in[12];
  const float* Ws2  = (const float*)d_in[13];
  const float* bs2  = (const float*)d_in[14];
  const float* mix  = (const float*)d_in[15];
  const int N  = in_sizes[2];
  const int NT = N / 16;

  char* ws = (char*)d_ws;
  size_t off = 0;
  auto take = [&](size_t b) { size_t r = off; off = (off + b + 255) & ~(size_t)255; return r; };
  size_t o_wp1 = take(64  * 128 * 2);
  size_t o_wpg = take(256 * 128 * 2);
  size_t o_wps = take(256 * 128 * 2);
  size_t o_zero = off;
  size_t o_b1s = take(512), o_b1q = take(512), o_b2s = take(512), o_b2q = take(512);
  size_t o_cnt = take(1024);
  size_t o_gsum = take(256 * 128 * 4);
  size_t o_gsq  = take(256 * 128 * 4);
  int zn = (int)((off - o_zero) / 4);
  size_t o_sc1 = take(512), o_sh1 = take(512), o_sc2 = take(512), o_sh2 = take(512);
  size_t o_seg = take(257 * 4);
  size_t o_smax = take(1024), o_sden = take(1024);
  size_t o_y1z = take((size_t)N * 128 * 2);
  size_t o_hf  = take((size_t)N * 128 * 2);
  size_t o_log = take((size_t)N * 4);

  k_zero<<<256, 256, 0, stream>>>((float*)(ws + o_zero), zn);
  k_swz<<<(2 * 4096 + 255) / 256, 256, 0, stream>>>(W1,  (__bf16*)(ws + o_wp1), 2);
  k_swz<<<(8 * 4096 + 255) / 256, 256, 0, stream>>>(Wg,  (__bf16*)(ws + o_wpg), 8);
  k_swz<<<(8 * 4096 + 255) / 256, 256, 0, stream>>>(Ws1, (__bf16*)(ws + o_wps), 8);
  k_gstats<<<(N + 63) / 64, 128, 0, stream>>>(h, batch, (float*)(ws + o_gsum),
                                              (float*)(ws + o_gsq), (float*)(ws + o_cnt), N);
  k_gemm1<<<1024, 256, 0, stream>>>(sub, (const __bf16*)(ws + o_wp1), b1,
                                    (__bf16*)(ws + o_y1z),
                                    (float*)(ws + o_b1s), (float*)(ws + o_b1q), NT);
  k_fin1<<<1, 256, 0, stream>>>(g1, be1, (const float*)(ws + o_b1s), (const float*)(ws + o_b1q),
                                (float*)(ws + o_sc1), (float*)(ws + o_sh1),
                                (float*)(ws + o_gsum), (float*)(ws + o_gsq),
                                (const float*)(ws + o_cnt), (int*)(ws + o_seg), (float)N);
  k_fused<<<1200, 256, 0, stream>>>(h, batch, (__bf16*)(ws + o_y1z), (__bf16*)(ws + o_hf),
                                    (const __bf16*)(ws + o_wpg), (const __bf16*)(ws + o_wps),
                                    bg, bs1, (const float*)(ws + o_sc1), (const float*)(ws + o_sh1),
                                    (const float*)(ws + o_gsum), (const float*)(ws + o_gsq),
                                    (float*)(ws + o_b2s), (float*)(ws + o_b2q), NT);
  k_fin2<<<1, 128, 0, stream>>>(g2, be2, (const float*)(ws + o_b2s), (const float*)(ws + o_b2q),
                                (float*)(ws + o_sc2), (float*)(ws + o_sh2), (float)N);
  k_logits<<<2048, 256, 0, stream>>>((const __bf16*)(ws + o_y1z), (const float*)(ws + o_sc2),
                                     (const float*)(ws + o_sh2), Ws2, bs2,
                                     (float*)(ws + o_log), N);
  k_seg<<<256, 256, 0, stream>>>((const float*)(ws + o_log), (const int*)(ws + o_seg),
                                 (float*)(ws + o_smax), (float*)(ws + o_sden));
  k_out<<<256, 128, 0, stream>>>((const __bf16*)(ws + o_hf), (const float*)(ws + o_log),
                                 (const int*)(ws + o_seg), (const float*)(ws + o_smax),
                                 (const float*)(ws + o_sden), (const float*)(ws + o_gsum),
                                 mix, (float*)d_out);
}
